// RNNClassifier_83837761618358
// MI455X (gfx1250) — matrix operand order repaired and checked by reading, not compiled, not run on hardware
//
#include <hip/hip_runtime.h>
#include <hip/hip_bf16.h>

typedef __bf16 bf16;
typedef __attribute__((ext_vector_type(16))) __bf16 v16bf;
typedef __attribute__((ext_vector_type(8)))  __bf16 v8bf;
typedef __attribute__((ext_vector_type(8)))  float  v8f;

// ---- problem sizes ----
constexpr int kB = 256;    // batch
constexpr int kT = 512;    // sequence length
constexpr int kE = 128;    // embedding dim
constexpr int kC = 256;    // 2E (conv output channels)
constexpr int kH = 256;    // LSTM hidden
constexpr int kL = 31;     // classes
constexpr int kG = 4 * kH; // 1024 gate outputs
constexpr int kK = kC + kH;// 512 combined GEMM K for LSTM

// ---- workspace layout (bytes) ----
constexpr size_t SZ_EMB  = (size_t)kB * kC * kT * 4;  // fp32 embeddings (B,2E,T) pre-BN
constexpr size_t SZ_X    = (size_t)kT * kB * kC * 2;  // bf16 act (T,B,2E)
constexpr size_t SZ_WT   = (size_t)kG * kK * 2;       // bf16 Wt (1024 x 512) = [w_ih|w_hh]
constexpr size_t SZ_W1R  = (size_t)kE * (kE * 3) * 2; // bf16 reordered conv1 weights
constexpr size_t SZ_W2R  = (size_t)kE * (kE * 5) * 2; // bf16 reordered conv2 weights
constexpr size_t OFF_EMB   = 0;
constexpr size_t OFF_X     = OFF_EMB + SZ_EMB;
constexpr size_t OFF_WT    = OFF_X + SZ_X;
constexpr size_t OFF_W1R   = OFF_WT + SZ_WT;
constexpr size_t OFF_W2R   = OFF_W1R + SZ_W1R;
constexpr size_t OFF_BIASC = OFF_W2R + SZ_W2R;        // 1024 f32 (b_ih+b_hh)
constexpr size_t OFF_SUM   = OFF_BIASC + 4096;        // 256 f32
constexpr size_t OFF_SUMSQ = OFF_SUM + 1024;          // 256 f32
constexpr size_t OFF_SCALE = OFF_SUMSQ + 1024;        // 256 f32
constexpr size_t OFF_SHIFT = OFF_SCALE + 1024;        // 256 f32
constexpr size_t OFF_HF    = OFF_SHIFT + 1024;        // fp32 h (B,H)
constexpr size_t OFF_CF    = OFF_HF + (size_t)kB * kH * 4;  // fp32 c (B,H)
constexpr size_t OFF_HBF0  = OFF_CF + (size_t)kB * kH * 4;  // bf16 h ping
constexpr size_t OFF_HBF1  = OFF_HBF0 + (size_t)kB * kH * 2;// bf16 h pong
constexpr size_t OFF_END   = OFF_HBF1 + (size_t)kB * kH * 2;

// ---- WMMA fragment helpers (wave32, 16x16x32 bf16; contiguous k = 16*(lane/16) + j in A and B) ----
__device__ __forceinline__ v16bf make16(v8bf lo, v8bf hi) {
  v16bf r;
#pragma unroll
  for (int j = 0; j < 8; ++j) { r[j] = lo[j]; r[8 + j] = hi[j]; }
  return r;
}
// A: lane holds row = lane%16; halves j(0..15)=A[row][kb+kg*16+j]
__device__ __forceinline__ v16bf load_afrag(const bf16* rowk /* &A[row][kb] */, int kgroup) {
  const v8bf* p0 = reinterpret_cast<const v8bf*>(rowk + kgroup * 16);
  const v8bf* p1 = reinterpret_cast<const v8bf*>(rowk + kgroup * 16 + 8);
  return make16(*p0, *p1);
}
// B: lane holds col = lane%16; halves j = B[kb + (lane/16)*16 + j][col] -> 16 contiguous
// when B is stored transposed (col-major as Wt[col][k]).
__device__ __forceinline__ v16bf load_bfrag16(const bf16* p /* 16 contiguous halves */) {
  const v8bf* q = reinterpret_cast<const v8bf*>(p);
  return make16(q[0], q[1]);
}
__device__ __forceinline__ float sigm(float x) { return 1.f / (1.f + __expf(-x)); }

// CDNA5 async copy: 16 bytes global -> LDS, tracked by ASYNCcnt (GVS addressing).
__device__ __forceinline__ void async_copy16(unsigned lds_addr, unsigned voff,
                                             const void* sbase) {
  asm volatile("global_load_async_to_lds_b128 %0, %1, %2"
               :: "v"(lds_addr), "v"(voff), "s"(sbase)
               : "memory");
}
__device__ __forceinline__ void async_wait0() {
  asm volatile("s_wait_asynccnt 0x0" ::: "memory");
}

// ================= K1: weight prep (Wt, biasc, reordered conv weights) =================
__global__ void prep_weights_kernel(const float* __restrict__ w_ih, const float* __restrict__ w_hh,
                                    const float* __restrict__ b_ih, const float* __restrict__ b_hh,
                                    const float* __restrict__ c1w, const float* __restrict__ c2w,
                                    bf16* __restrict__ Wt, float* __restrict__ biasc,
                                    bf16* __restrict__ W1r, bf16* __restrict__ W2r) {
  const size_t NWT = (size_t)kG * kK;            // 524288
  const size_t NB  = NWT + kG;                   // +1024
  const size_t N1  = NB + (size_t)kE * kE * 3;   // +49152
  const size_t N2  = N1 + (size_t)kE * kE * 5;   // +81920
  size_t i = (size_t)blockIdx.x * blockDim.x + threadIdx.x;
  if (i < NWT) {
    int g = (int)(i / kK), k = (int)(i % kK);
    float v = (k < kC) ? w_ih[(size_t)g * kC + k] : w_hh[(size_t)g * kH + (k - kC)];
    Wt[i] = (bf16)v;
  } else if (i < NB) {
    int g = (int)(i - NWT);
    biasc[g] = b_ih[g] + b_hh[g];
  } else if (i < N1) {
    size_t idx = i - NB;                 // c*384 + kk ; kk = k*128 + ci
    int c = (int)(idx / 384), kk = (int)(idx % 384);
    int k = kk >> 7, ci = kk & 127;
    W1r[idx] = (bf16)c1w[((size_t)c * kE + ci) * 3 + k];
  } else if (i < N2) {
    size_t idx = i - N1;                 // c*640 + kk ; kk = k*128 + ci
    int c = (int)(idx / 640), kk = (int)(idx % 640);
    int k = kk >> 7, ci = kk & 127;
    W2r[idx] = (bf16)c2w[((size_t)c * kE + ci) * 5 + k];
  }
}

// ================= K2: embedding gather + conv1(K=3) + conv2(K=5) via WMMA ==============
// block = (b, t-tile of 16); 256 threads = 8 waves; wave w computes out-channel tiles
// {w} (conv1) and {w+8} (conv2). A = reordered weights, B = im2col from LDS.
__global__ void __launch_bounds__(256)
conv_embed_kernel(const int* __restrict__ x, const float* __restrict__ etab,
                  const bf16* __restrict__ W1r, const bf16* __restrict__ W2r,
                  const float* __restrict__ c1b, const float* __restrict__ c2b,
                  float* __restrict__ emb) {
  constexpr int LROWS = 20;   // t0-2 .. t0+17
  constexpr int LSTR  = 136;  // 128 + 8 pad halves
  __shared__ bf16 smem[LROWS * LSTR];
  const int t0 = blockIdx.x * 16;
  const int b  = blockIdx.y;
  // fill LDS: rows t0-2..t0+17, 128 channels each, bf16, zero-padded outside [0,T)
  for (int e = threadIdx.x; e < LROWS * kE; e += 256) {
    int r = e >> 7, ch = e & 127;
    int t = t0 - 2 + r;
    float v = 0.f;
    if (t >= 0 && t < kT) {
      int tok = x[(size_t)b * kT + t];
      v = etab[(size_t)tok * kE + ch];
    }
    smem[r * LSTR + ch] = (bf16)v;
  }
  __syncthreads();

  const int wid   = threadIdx.x >> 5;
  const int lane  = threadIdx.x & 31;
  const int lrow  = lane & 15;
  const int khalf = lane >> 4;

#pragma unroll
  for (int phase = 0; phase < 2; ++phase) {
    const bf16* Wr = phase ? W2r : W1r;
    const int Kd = phase ? 640 : 384;
    const int tshift = phase ? 0 : 1;            // 2 - pad
    const int cbase = phase ? (128 + wid * 16) : (wid * 16);
    v8f acc = {};
    const bf16* arow = Wr + (size_t)((cbase & 127) + lrow) * Kd;
    for (int ks = 0; ks < Kd / 32; ++ks) {
      const int kb = ks * 32;
      v16bf a = load_afrag(arow + kb, khalf);
      const int kk0 = kb + khalf * 16;
      const int kconv = kk0 >> 7;                // which conv tap
      const int i0 = kk0 & 127;                  // input channel base
      const int tl = lrow + kconv + tshift;      // LDS row
      v16bf bf = load_bfrag16(&smem[tl * LSTR + i0]);
      acc = __builtin_amdgcn_wmma_f32_16x16x32_bf16(false, a, false, bf, (short)0, acc,
                                                    false, false);
    }
    // D: lane holds rows c = cbase + vg + khalf*8, col t = t0 + lrow
    float* outp = emb + ((size_t)b * kC + cbase + khalf * 8) * kT + t0 + lrow;
#pragma unroll
    for (int vg = 0; vg < 8; ++vg) {
      int c = cbase + khalf * 8 + vg;
      float bias = phase ? c2b[c - 128] : c1b[c];
      outp[(size_t)vg * kT] = acc[vg] + bias;
    }
  }
}

// ================= K3: BN batch statistics (per-channel sum / sumsq) ====================
__global__ void __launch_bounds__(256)
bn_stats_kernel(const float* __restrict__ emb, float* __restrict__ sum, float* __restrict__ sumsq) {
  __shared__ float ss[256], s2[256];
  const int c = blockIdx.x, tid = threadIdx.x;
  float a = 0.f, a2 = 0.f;
  for (int idx = tid; idx < kB * kT; idx += 256) {
    int b = idx >> 9, t = idx & 511;
    float v = emb[((size_t)b * kC + c) * kT + t];
    a += v; a2 += v * v;
  }
  ss[tid] = a; s2[tid] = a2;
  __syncthreads();
  for (int off = 128; off > 0; off >>= 1) {
    if (tid < off) { ss[tid] += ss[tid + off]; s2[tid] += s2[tid + off]; }
    __syncthreads();
  }
  if (tid == 0) { sum[c] = ss[0]; sumsq[c] = s2[0]; }
}

__global__ void bn_scale_kernel(const float* __restrict__ sum, const float* __restrict__ sumsq,
                                const float* __restrict__ gamma, const float* __restrict__ beta,
                                float* __restrict__ scale, float* __restrict__ shift) {
  int c = threadIdx.x;
  const float inv_n = 1.f / (float)(kB * kT);
  float mean = sum[c] * inv_n;
  float var = sumsq[c] * inv_n - mean * mean;
  float sc = gamma[c] * rsqrtf(var + 1e-5f);
  scale[c] = sc;
  shift[c] = beta[c] - mean * sc;
}

// ================= K4: BN + ReLU + pack to bf16 X (T,B,2E) ==============================
__global__ void __launch_bounds__(256)
bn_relu_pack_kernel(const float* __restrict__ emb, const float* __restrict__ scale,
                    const float* __restrict__ shift, bf16* __restrict__ X) {
  const int t0 = blockIdx.x * 16;
  const int b  = blockIdx.y;
  const int c  = threadIdx.x;
  const float sc = scale[c], sh = shift[c];
  const float* src = emb + ((size_t)b * kC + c) * kT + t0;
#pragma unroll
  for (int i = 0; i < 16; ++i) {
    float v = fmaxf(src[i] * sc + sh, 0.f);
    X[((size_t)(t0 + i) * kB + b) * kC + c] = (bf16)v;   // coalesced across c
  }
}

// ================= K5: one LSTM time step (WMMA gate GEMM + cell update) ================
// grid = (B/16, H/16); block = 128 (4 waves); wave q computes gate q's 16x16 tile with
// K = 512 over A = [x_t | h_prev] staged in LDS via async copies; B = Wt (transposed
// weights, 16 contiguous K halves per lane). h is double-buffered across launches to
// avoid the cross-block RAW race on the recurrent input.
constexpr int ASTR = 520;  // A-tile LDS row stride in halves (1040 B: 16B-aligned, bank-skewed)
__global__ void __launch_bounds__(128)
lstm_step_kernel(const bf16* __restrict__ X, const bf16* __restrict__ Wt,
                 const float* __restrict__ biasc, const int* __restrict__ lengths,
                 float* __restrict__ Hf, float* __restrict__ Cf,
                 const bf16* __restrict__ Hprev, bf16* __restrict__ Hnext, int t) {
  __shared__ bf16 smemA[16 * ASTR];
  __shared__ float g4[4][16][16];
  const int mbase = blockIdx.x * 16;     // batch rows
  const int hbase = blockIdx.y * 16;     // hidden cols
  const int q     = threadIdx.x >> 5;    // gate: 0=i 1=f 2=g 3=o
  const int lane  = threadIdx.x & 31;
  const int lrow  = lane & 15;
  const int khalf = lane >> 4;

  // ---- async-stage A tile: row r = [ X[t][mbase+r][0:256] | Hprev[mbase+r][0:256] ] ----
  const unsigned lbase = (unsigned)(uintptr_t)smemA;      // low 32 bits = LDS offset
  const unsigned xtoff = (unsigned)((size_t)t * kB * kC * 2);  // byte offset of x_t in X
#pragma unroll
  for (int i = 0; i < 4; ++i) {
    const int chunk = threadIdx.x + i * 128;              // 0..511 (16B chunks)
    const int r  = chunk >> 5;                            // row 0..15
    const int cb = (chunk & 31) * 16;                     // byte offset within 512B half-row
    async_copy16(lbase + r * (ASTR * 2) + cb,
                 xtoff + (unsigned)((mbase + r) * kC * 2 + cb), X);
    async_copy16(lbase + r * (ASTR * 2) + 512 + cb,
                 (unsigned)((mbase + r) * kH * 2 + cb), Hprev);
  }
  async_wait0();
  __syncthreads();

  const int ncol = q * kH + hbase + lrow;               // gate output index
  const bf16* wrow = Wt + (size_t)ncol * kK;            // Wt[col][k], contiguous in k
  const bf16* arow = smemA + lrow * ASTR;               // A row in LDS (512 halves)

  v8f acc = {};
#pragma unroll
  for (int ks = 0; ks < kK / 32; ++ks) {
    const int kb = ks * 32;
    v16bf a = load_afrag(arow + kb, khalf);             // ds_load_b128 x2
    v16bf bf = load_bfrag16(wrow + kb + khalf * 16);    // global_load_b128 x2
    acc = __builtin_amdgcn_wmma_f32_16x16x32_bf16(false, a, false, bf, (short)0, acc,
                                                  false, false);
  }
  const float bias = biasc[ncol];
#pragma unroll
  for (int vg = 0; vg < 8; ++vg)
    g4[q][khalf * 8 + vg][lrow] = acc[vg] + bias;
  __syncthreads();

  for (int e = threadIdx.x; e < 256; e += 128) {
    const int row = e >> 4, col = e & 15;
    const int b = mbase + row, u = hbase + col;
    const size_t idx = (size_t)b * kH + u;
    if (t < lengths[b]) {
      const float I = g4[0][row][col], F = g4[1][row][col];
      const float G = g4[2][row][col], O = g4[3][row][col];
      const float c_old = Cf[idx];
      const float c_new = sigm(F) * c_old + sigm(I) * tanhf(G);
      const float h_new = sigm(O) * tanhf(c_new);
      Cf[idx] = c_new;
      Hf[idx] = h_new;
      Hnext[idx] = (bf16)h_new;
    } else {
      Hnext[idx] = Hprev[idx];   // carry state forward in the ping-pong buffer
    }
  }
}

// ================= K6: attention (pre-BN embeddings x hn) + logits ======================
__global__ void __launch_bounds__(256)
attn_logits_kernel(const float* __restrict__ emb, const float* __restrict__ Hf,
                   const float* __restrict__ lin_w, const float* __restrict__ lin_b,
                   float* __restrict__ out) {
  __shared__ float hs[kH];
  __shared__ float at[kT];
  const int b = blockIdx.x, tid = threadIdx.x;
  hs[tid] = Hf[(size_t)b * kH + tid];
  __syncthreads();
  // attn[b][t] = (emb[b][:,t] . hn[b]) / 16
  for (int t = tid; t < kT; t += 256) {
    const float* eb = emb + (size_t)b * kC * kT + t;
    float a = 0.f;
    for (int c = 0; c < kC; ++c) a += eb[(size_t)c * kT] * hs[c];
    at[t] = a * (1.f / 16.f);
  }
  __syncthreads();
  // attn_out[b][c] = sum_t attn[b][t] * emb[b][c][t]
  {
    const float* eb = emb + ((size_t)b * kC + tid) * kT;
    float a = 0.f;
    for (int t = 0; t < kT; ++t) a += at[t] * eb[t];
    out[(size_t)kB * kL + (size_t)b * kC + tid] = a;
  }
  // logits[b][l]
  if (tid < kL) {
    float a = lin_b[tid];
    for (int h = 0; h < kH; ++h) a += hs[h] * lin_w[(size_t)tid * kH + h];
    out[(size_t)b * kL + tid] = a;
  }
}

// =======================================================================================
extern "C" void kernel_launch(void* const* d_in, const int* in_sizes, int n_in,
                              void* d_out, int out_size, void* d_ws, size_t ws_size,
                              hipStream_t stream) {
  (void)in_sizes; (void)n_in; (void)out_size; (void)ws_size;
  const int*   x        = (const int*)d_in[0];
  const int*   lengths  = (const int*)d_in[1];
  const float* etab     = (const float*)d_in[2];
  const float* conv1_w  = (const float*)d_in[3];
  const float* conv1_b  = (const float*)d_in[4];
  const float* conv2_w  = (const float*)d_in[5];
  const float* conv2_b  = (const float*)d_in[6];
  const float* bn_gamma = (const float*)d_in[7];
  const float* bn_beta  = (const float*)d_in[8];
  const float* w_ih     = (const float*)d_in[9];
  const float* w_hh     = (const float*)d_in[10];
  const float* b_ih     = (const float*)d_in[11];
  const float* b_hh     = (const float*)d_in[12];
  const float* lin_w    = (const float*)d_in[13];
  const float* lin_b    = (const float*)d_in[14];
  float* out = (float*)d_out;

  char* ws = (char*)d_ws;
  float* emb   = (float*)(ws + OFF_EMB);
  bf16*  X     = (bf16*)(ws + OFF_X);
  bf16*  Wt    = (bf16*)(ws + OFF_WT);
  bf16*  W1r   = (bf16*)(ws + OFF_W1R);
  bf16*  W2r   = (bf16*)(ws + OFF_W2R);
  float* biasc = (float*)(ws + OFF_BIASC);
  float* sum   = (float*)(ws + OFF_SUM);
  float* sumsq = (float*)(ws + OFF_SUMSQ);
  float* scale = (float*)(ws + OFF_SCALE);
  float* shift = (float*)(ws + OFF_SHIFT);
  float* Hf    = (float*)(ws + OFF_HF);
  float* Cf    = (float*)(ws + OFF_CF);
  bf16*  Hb0   = (bf16*)(ws + OFF_HBF0);
  bf16*  Hb1   = (bf16*)(ws + OFF_HBF1);

  // zero LSTM state (graph-capturable)
  hipMemsetAsync(Hf, 0, (size_t)kB * kH * 4, stream);
  hipMemsetAsync(Cf, 0, (size_t)kB * kH * 4, stream);
  hipMemsetAsync(Hb0, 0, (size_t)kB * kH * 2, stream);
  hipMemsetAsync(Hb1, 0, (size_t)kB * kH * 2, stream);

  // weight prep
  {
    const size_t N = (size_t)kG * kK + kG + (size_t)kE * kE * 3 + (size_t)kE * kE * 5;
    prep_weights_kernel<<<(unsigned)((N + 255) / 256), 256, 0, stream>>>(
        w_ih, w_hh, b_ih, b_hh, conv1_w, conv2_w, Wt, biasc, W1r, W2r);
  }
  // embedding + convs (WMMA)
  conv_embed_kernel<<<dim3(kT / 16, kB), 256, 0, stream>>>(x, etab, W1r, W2r, conv1_b,
                                                           conv2_b, emb);
  // batchnorm stats + fold
  bn_stats_kernel<<<kC, 256, 0, stream>>>(emb, sum, sumsq);
  bn_scale_kernel<<<1, kC, 0, stream>>>(sum, sumsq, bn_gamma, bn_beta, scale, shift);
  // BN + ReLU + bf16 pack to (T,B,2E)
  bn_relu_pack_kernel<<<dim3(kT / 16, kB), 256, 0, stream>>>(emb, scale, shift, X);
  // sequential LSTM: 512 stream-ordered launches; h ping-pongs between launches
  for (int t = 0; t < kT; ++t) {
    const bf16* Hprev = (t & 1) ? Hb1 : Hb0;
    bf16*       Hnext = (t & 1) ? Hb0 : Hb1;
    lstm_step_kernel<<<dim3(kB / 16, kH / 16), 128, 0, stream>>>(X, Wt, biasc, lengths,
                                                                 Hf, Cf, Hprev, Hnext, t);
  }
  // attention + classifier head
  attn_logits_kernel<<<kB, 256, 0, stream>>>(emb, Hf, lin_w, lin_b, out);
}
